// Attention_52415780880456
// MI455X (gfx1250) — matrix operand order repaired and checked by reading, not compiled, not run on hardware
//
#include <hip/hip_runtime.h>

// ---------------------------------------------------------------------------
// Attention for B=64, C=8, H=W=256, CQ=1 on gfx1250 (MI455X).
//   K1: fused 1x1-conv q/k/v  (v stored bf16 TRANSPOSED for WMMA-B layout)
//   K2: scores = Q @ K^T      via V_WMMA_F32_16X16X4_F32, 32x64 block/wave
//   K3: row softmax           (wave-per-row, wave32 shfl reductions) -> bf16
//   K4: out = attn @ V        via V_WMMA_F32_16X16X32_BF16, 32x64 block/wave
// Intermediates (~126 MB) are L2-resident (192 MB L2); 2D register blocking
// (2 i-tiles x 4 n-tiles = 8 accumulators/wave) amortizes BOTH operand
// fragments, so the GEMMs run against the WMMA pipes, not the load path
// (K4: 1.5 b128-loads per WMMA; K2: 0.75 b64-loads per WMMA).
// ---------------------------------------------------------------------------

typedef __attribute__((ext_vector_type(2)))  float  v2f;
typedef __attribute__((ext_vector_type(8)))  float  v8f;
typedef __attribute__((ext_vector_type(8)))  __bf16 v8bf;
typedef __attribute__((ext_vector_type(16))) __bf16 v16bf;

#define NB 64
#define NC 8
#define NH 256
#define NW 256
static constexpr size_t NPIX = (size_t)NH * NW;       // 65536
static constexpr size_t QK_ELEMS = (size_t)NB * NPIX; // 4194304

// ---------------------------------------------------------------------------
// Kernel 1: per 16x16 spatial tile, compute q,k (fp32, row-major) and v
// (bf16), writing v transposed vt[(b*8+c)][w][j] so K4's B-matrix loads are
// contiguous. LDS tile used for the in-block transpose (conflict-padded).
// ---------------------------------------------------------------------------
__global__ __launch_bounds__(256) void qkv_kernel(
    const float* __restrict__ x,
    const float* __restrict__ w1, const float* __restrict__ b1,
    const float* __restrict__ w2, const float* __restrict__ b2,
    const float* __restrict__ w3, const float* __restrict__ b3,
    float* __restrict__ qo, float* __restrict__ ko, __bf16* __restrict__ vt)
{
    __shared__ float ldsv[NC][16][17];

    const int tile = blockIdx.x;          // NB*256 tiles
    const int b    = tile >> 8;
    const int rem  = tile & 255;
    const int j0   = (rem >> 4) << 4;     // spatial h tile
    const int w0   = (rem & 15) << 4;     // spatial w tile
    const int tx   = threadIdx.x & 15;    // w offset
    const int ty   = threadIdx.x >> 4;    // j offset
    const size_t pix = (size_t)(j0 + ty) * NW + (w0 + tx);
    const float* xb  = x + (size_t)b * NC * NPIX;

    float xc[NC];
#pragma unroll
    for (int c = 0; c < NC; ++c) xc[c] = xb[c * NPIX + pix];

    float qv = b1[0], kv = b2[0];
#pragma unroll
    for (int c = 0; c < NC; ++c) { qv += w1[c] * xc[c]; kv += w2[c] * xc[c]; }
    qo[(size_t)b * NPIX + pix] = qv;
    ko[(size_t)b * NPIX + pix] = kv;

#pragma unroll
    for (int o = 0; o < NC; ++o) {
        float vv = b3[o];
#pragma unroll
        for (int c = 0; c < NC; ++c) vv += w3[o * NC + c] * xc[c];
        ldsv[o][ty][tx] = vv;
    }
    __syncthreads();
    // transposed store: out element is v[c][j0+tx][w0+ty] = ldsv[c][tx][ty]
#pragma unroll
    for (int o = 0; o < NC; ++o) {
        vt[(((size_t)(b * NC + o)) * NW + (w0 + ty)) * NH + (j0 + tx)] =
            (__bf16)ldsv[o][tx][ty];
    }
}

// ---------------------------------------------------------------------------
// Kernel 2: scores[b,i,j] = sum_w q[b,i,w]*k[b,j,w].
// One wave per 32(i) x 64(j) block: per k-step, 2 Q (A) fragments and 4 K^T
// (B) fragments feed 8 WMMAs (2x4 accumulator grid).
// fp32 A 16x4 layout: lane l holds row (l&15), K = 2*(l>>4) + {0,1}.
// fp32 B 4x16 layout: lane l holds col (l&15), K = 2*(l>>4) + {0,1} of K^T,
// i.e. a float2 from k-row (jbase + (l&15)) — both operands stay row-major.
// ---------------------------------------------------------------------------
__global__ __launch_bounds__(256) void scores_kernel(
    const float* __restrict__ q, const float* __restrict__ k,
    float* __restrict__ sc)
{
    const int wave = (blockIdx.x << 3) + (threadIdx.x >> 5); // NB*8*4 waves
    const int lane = threadIdx.x & 31;
    const int b    = wave >> 5;          // 32 waves per batch
    const int rem  = wave & 31;
    const int i0   = (rem >> 2) << 5;    // 8 i-blocks of 32
    const int j0   = (rem & 3) << 6;     // 4 j-blocks of 64
    const int m    = lane & 15;
    const int hl   = lane >> 4;

    const float* qrow = q + (size_t)b * NPIX + (size_t)(i0 + m) * NW;
    const float* krow = k + (size_t)b * NPIX + (size_t)(j0 + m) * NW;

    v8f acc[2][4] = {};
#if defined(__gfx1250__) && __has_builtin(__builtin_amdgcn_wmma_f32_16x16x4_f32)
#pragma unroll 4
    for (int kk = 0; kk < NW; kk += 4) {
        v2f av[2];
#pragma unroll
        for (int u = 0; u < 2; ++u)
            av[u] = *(const v2f*)(qrow + u * (16 * NW) + kk + hl * 2);
#pragma unroll
        for (int t = 0; t < 4; ++t) {
            const v2f bv = *(const v2f*)(krow + t * (16 * NW) + kk + hl * 2);
#pragma unroll
            for (int u = 0; u < 2; ++u)
                acc[u][t] = __builtin_amdgcn_wmma_f32_16x16x4_f32(
                    /*neg_a=*/false, av[u], /*neg_b=*/false, bv,
                    /*c_mod=*/(short)0, acc[u][t],
                    /*reuse_a=*/false, /*reuse_b=*/false);
        }
    }
#else
    // scalar fallback matching the C/D VGPR layout
    for (int u = 0; u < 2; ++u)
        for (int t = 0; t < 4; ++t)
            for (int r = 0; r < 8; ++r) {
                const float* qr =
                    q + (size_t)b * NPIX + (size_t)(i0 + u * 16 + hl * 8 + r) * NW;
                const float* kr =
                    k + (size_t)b * NPIX + (size_t)(j0 + t * 16 + m) * NW;
                float s = 0.f;
                for (int kk = 0; kk < NW; ++kk) s += qr[kk] * kr[kk];
                acc[u][t][r] = s;
            }
#endif
    // D layout: VGPR r -> row i0 + u*16 + hl*8 + r, col j0 + t*16 + (lane&15)
#pragma unroll
    for (int u = 0; u < 2; ++u)
#pragma unroll
        for (int t = 0; t < 4; ++t)
#pragma unroll
            for (int r = 0; r < 8; ++r)
                sc[(size_t)b * NPIX + (size_t)(i0 + u * 16 + hl * 8 + r) * NW +
                   (j0 + t * 16 + m)] = acc[u][t][r];
}

// ---------------------------------------------------------------------------
// Kernel 3: softmax over each scores row (256 elems), wave-per-row.
// fp32 max/exp/sum with wave32 shfl_xor reductions; emit attn as bf16.
// ---------------------------------------------------------------------------
__global__ __launch_bounds__(256) void softmax_kernel(
    const float* __restrict__ sc, __bf16* __restrict__ attn)
{
    const int row  = (blockIdx.x << 3) + (threadIdx.x >> 5); // B*H rows
    const int lane = threadIdx.x & 31;
    const float* p = sc + (size_t)row * NH;

    float v[8];
#pragma unroll
    for (int t = 0; t < 8; ++t) v[t] = p[lane + 32 * t];

    float mx = v[0];
#pragma unroll
    for (int t = 1; t < 8; ++t) mx = fmaxf(mx, v[t]);
#pragma unroll
    for (int s = 16; s > 0; s >>= 1) mx = fmaxf(mx, __shfl_xor(mx, s, 32));

    float sum = 0.f;
#pragma unroll
    for (int t = 0; t < 8; ++t) { v[t] = __expf(v[t] - mx); sum += v[t]; }
#pragma unroll
    for (int s = 16; s > 0; s >>= 1) sum += __shfl_xor(sum, s, 32);

    const float inv = __frcp_rn(sum);
    __bf16* o = attn + (size_t)row * NH;
#pragma unroll
    for (int t = 0; t < 8; ++t) o[lane + 32 * t] = (__bf16)(v[t] * inv);
}

// ---------------------------------------------------------------------------
// Kernel 4: out[b,c] = attn[b] (256x256) @ V[b,c] (256x256), bf16 WMMA.
// One wave per 32(i) x 64(w) block: per 32-wide k-step, 2 attn (A) fragments
// and 4 V (B) fragments feed 8 WMMAs (2x4 accumulator grid). K loop = 8 steps.
// A (attn, 16x32 bf16): lane l row (l&15); elems 0-7 = K hl*8+{0..7},
//   elems 8-15 = K 16+hl*8+{0..7}  -> two contiguous v8bf (b128) loads.
// B (V, 32x16 bf16): lane l col (l&15); elems 0-7 = V-row hl*8+{0..7}, elems 8-15 = 16+hl*8+{0..7}
//   -> with vt[w][j] storage these are two runs of 8 contiguous bf16 (two b128 loads).
// ---------------------------------------------------------------------------
__global__ __launch_bounds__(256) void pv_kernel(
    const __bf16* __restrict__ attn, const __bf16* __restrict__ vt,
    float* __restrict__ out)
{
    const int wave = (blockIdx.x << 3) + (threadIdx.x >> 5); // NB*NC*8*4 waves
    const int lane = threadIdx.x & 31;
    const int bc   = wave >> 5;          // b*8 + c   (32 waves per bc)
    const int b    = bc >> 3;
    const int rem  = wave & 31;
    const int i0   = (rem >> 2) << 5;    // 8 i-blocks of 32
    const int w0   = (rem & 3) << 6;     // 4 w-blocks of 64
    const int m    = lane & 15;
    const int hl   = lane >> 4;

    const __bf16* arow = attn + ((size_t)b * NH + (i0 + m)) * NH;
    const __bf16* vrow = vt + ((size_t)bc * NW + (w0 + m)) * NH;

    v8f acc[2][4] = {};
#pragma unroll
    for (int kk = 0; kk < NH; kk += 32) {
        union { v16bf v; v8bf h[2]; } A[2];
#pragma unroll
        for (int u = 0; u < 2; ++u) {
            A[u].h[0] = *(const v8bf*)(arow + u * (16 * NH) + kk + hl * 8);
            A[u].h[1] = *(const v8bf*)(arow + u * (16 * NH) + kk + 16 + hl * 8);
        }
#pragma unroll
        for (int t = 0; t < 4; ++t) {
            union { v16bf v; v8bf h[2]; } Bm;
            Bm.h[0] = *(const v8bf*)(vrow + t * (16 * NH) + kk + hl * 8);
            Bm.h[1] = *(const v8bf*)(vrow + t * (16 * NH) + kk + 16 + hl * 8);
#pragma unroll
            for (int u = 0; u < 2; ++u)
                acc[u][t] = __builtin_amdgcn_wmma_f32_16x16x32_bf16(
                    /*neg_a=*/false, A[u].v, /*neg_b=*/false, Bm.v,
                    /*c_mod=*/(short)0, acc[u][t],
                    /*reuse_a=*/false, /*reuse_b=*/false);
        }
    }
    // D layout: VGPR r -> row i0 + u*16 + hl*8 + r, col w0 + t*16 + (lane&15)
#pragma unroll
    for (int u = 0; u < 2; ++u)
#pragma unroll
        for (int t = 0; t < 4; ++t)
#pragma unroll
            for (int r = 0; r < 8; ++r)
                out[(size_t)bc * NPIX + (size_t)(i0 + u * 16 + hl * 8 + r) * NW +
                    (w0 + t * 16 + m)] = acc[u][t][r];
}

// ---------------------------------------------------------------------------
extern "C" void kernel_launch(void* const* d_in, const int* in_sizes, int n_in,
                              void* d_out, int out_size, void* d_ws, size_t ws_size,
                              hipStream_t stream)
{
    (void)in_sizes; (void)n_in; (void)out_size; (void)ws_size;
    const float* x  = (const float*)d_in[0];
    const float* w1 = (const float*)d_in[1];
    const float* b1 = (const float*)d_in[2];
    const float* w2 = (const float*)d_in[3];
    const float* b2 = (const float*)d_in[4];
    const float* w3 = (const float*)d_in[5];
    const float* b3 = (const float*)d_in[6];
    float* out = (float*)d_out;

    // workspace carve-up (all regions 16B aligned): ~126 MB total
    char* ws = (char*)d_ws;
    float*  q    = (float*)ws;  ws += QK_ELEMS * sizeof(float);   // 16 MB
    float*  k    = (float*)ws;  ws += QK_ELEMS * sizeof(float);   // 16 MB
    float*  sc   = (float*)ws;  ws += QK_ELEMS * sizeof(float);   // 16 MB (B*H*H)
    __bf16* attn = (__bf16*)ws; ws += QK_ELEMS * sizeof(__bf16);  //  8 MB
    __bf16* vt   = (__bf16*)ws;                                   // 64 MB (B*C*H*W)

    // K1: 64*256 tiles of 16x16 pixels, 256 threads each
    qkv_kernel<<<NB * 256, 256, 0, stream>>>(x, w1, b1, w2, b2, w3, b3, q, k, vt);
    // K2: 64 * (8 i-blocks * 4 j-blocks) waves, 8 waves/block
    scores_kernel<<<(NB * 32) / 8, 256, 0, stream>>>(q, k, sc);
    // K3: 64*256 rows, 8 waves/block
    softmax_kernel<<<(NB * NH) / 8, 256, 0, stream>>>(sc, attn);
    // K4: 512 * (8 i-blocks * 4 w-blocks) waves, 8 waves/block
    pv_kernel<<<(NB * NC * 32) / 8, 256, 0, stream>>>(attn, vt, out);
}
